// MRN_50663434224498
// MI455X (gfx1250) — matrix operand order repaired and checked by reading, not compiled, not run on hardware
//
#include <hip/hip_runtime.h>
#include <hip/hip_bf16.h>
#include <math.h>

#define N_NODES   8192
#define E_EDGES   49152
#define ETOT      (E_EDGES + N_NODES)   /* 57344 */
#define H_HEADS   6
#define C_CH      512
#define F_IN      514
#define KPAD      544
#define KTILES    17                     /* KPAD/32 */
#define CN        3072                   /* H*C */
#define NT64N     (CN / 64)              /* 48 */
#define LRELU     0.2f
#define SELU_SCALE 1.0507009873554805f
#define SELU_ALPHA 1.6732632423543772f

typedef __attribute__((ext_vector_type(16))) __bf16 v16bf;
typedef __attribute__((ext_vector_type(8)))  float  v8f;
typedef __attribute__((ext_vector_type(4)))  unsigned int u32x4;
typedef int v4i_ __attribute__((vector_size(16)));   // matches builtin param spelling

union ABFrag { v16bf v; u32x4 u[2]; };

__device__ __forceinline__ unsigned short f2bf(float f) {
    unsigned u = __float_as_uint(f);
    unsigned r = u + 0x7FFFu + ((u >> 16) & 1u);   // round-to-nearest-even
    return (unsigned short)(r >> 16);
}

__device__ __forceinline__ float selu_f(float x) {
    return SELU_SCALE * (x > 0.f ? x : SELU_ALPHA * (__expf(x) - 1.f));
}

__device__ __forceinline__ void atomicMaxFloat(float* addr, float v) {
    if (!(v < 0.f)) atomicMax((int*)addr, __float_as_int(v));
    else            atomicMin((unsigned int*)addr, __float_as_uint(v));
}

// ---- CDNA5 async global->LDS copy (ASYNCcnt path), with portable fallbacks
__device__ __forceinline__ void async_copy_b128(const unsigned short* g, unsigned short* l) {
#if defined(__gfx1250__) && __has_builtin(__builtin_amdgcn_global_load_async_to_lds_b128)
    __builtin_amdgcn_global_load_async_to_lds_b128(
        (v4i_ __attribute__((address_space(1)))*)(g),
        (v4i_ __attribute__((address_space(3)))*)(l), 0, 0);
#else
    *(u32x4*)l = *(const u32x4*)g;
#endif
}

__device__ __forceinline__ void wait_async_all() {
#if defined(__gfx1250__) && __has_builtin(__builtin_amdgcn_s_wait_asynccnt)
    __builtin_amdgcn_s_wait_asynccnt(0);
#elif defined(__gfx1250__)
    asm volatile("s_wait_asynccnt 0" ::: "memory");
#endif
}

// ---------------------------------------------------------------- fill
__global__ void fill_f32_kernel(float* __restrict__ p, float v, int n) {
    int i = blockIdx.x * blockDim.x + threadIdx.x;
    if (i < n) p[i] = v;
}

// ---------------------------------------------------------------- coord init
__global__ void coord_init_kernel(const float* __restrict__ x, float* __restrict__ coord) {
    int n = blockIdx.x * blockDim.x + threadIdx.x;
    if (n < N_NODES) {
        coord[2 * n + 0] = x[7 * n + 0];
        coord[2 * n + 1] = x[7 * n + 1];
    }
}

// ---------------------------------------------------------------- initial linear + SELU
__global__ void init_hidden_kernel(const float* __restrict__ x,
                                   const float* __restrict__ local_feat,
                                   const float* __restrict__ conv_feat,
                                   const float* __restrict__ W_lin,
                                   const float* __restrict__ b_lin,
                                   float* __restrict__ hidden) {
    int idx = blockIdx.x * blockDim.x + threadIdx.x;
    if (idx >= N_NODES * C_CH) return;
    int n = idx / C_CH;
    int c = idx - n * C_CH;
    float s = b_lin[c];
    #pragma unroll
    for (int k = 0; k < 5; ++k)  s += x[n * 7 + 2 + k]      * W_lin[k * C_CH + c];
    #pragma unroll
    for (int k = 0; k < 16; ++k) s += local_feat[n * 16 + k] * W_lin[(5 + k) * C_CH + c];
    #pragma unroll
    for (int k = 0; k < 16; ++k) s += conv_feat[n * 16 + k]  * W_lin[(21 + k) * C_CH + c];
    hidden[idx] = selu_f(s);
}

// ---------------------------------------------------------------- pack W (F_IN x CN f32) into WMMA-B bf16 fragments
// Layout: one contiguous 4KB block per (nt64, kt):
// packed[(((nt64*KTILES + kt)*4 + j)*32 + lane)*16 + i]
//   = W[kt*32 + (lane>>4)*16 + i][(nt64*4 + j)*16 + (lane&15)]
__global__ void pack_w_kernel(const float* __restrict__ W, unsigned short* __restrict__ Bp) {
    int idx = blockIdx.x * blockDim.x + threadIdx.x;
    if (idx >= NT64N * KTILES * 4 * 32) return;
    int lane = idx & 31;
    int t    = idx >> 5;            // ((nt64*KTILES + kt)*4 + j)
    int j    = t & 3;
    int t2   = t >> 2;              // nt64*KTILES + kt
    int kt   = t2 % KTILES;
    int nt64 = t2 / KTILES;
    int col   = (nt64 * 4 + j) * 16 + (lane & 15);
    int kbase = kt * 32 + ((lane >> 4) << 4);
    unsigned short* out = Bp + (size_t)idx * 16;
    #pragma unroll
    for (int i = 0; i < 16; ++i) {
        int k = kbase + i;
        float v = (k < F_IN) ? W[(size_t)k * CN + col] : 0.f;
        out[i] = f2bf(v);
    }
}

// ---------------------------------------------------------------- build feat = [coord | hidden | zero-pad] bf16
__global__ void build_feat_kernel(const float* __restrict__ coord,
                                  const float* __restrict__ hidden,
                                  unsigned short* __restrict__ feat) {
    int idx = blockIdx.x * blockDim.x + threadIdx.x;
    if (idx >= N_NODES * KPAD) return;
    int n = idx / KPAD;
    int k = idx - n * KPAD;
    float v;
    if (k < 2)          v = coord[n * 2 + k];
    else if (k < F_IN)  v = hidden[(size_t)n * C_CH + (k - 2)];
    else                v = 0.f;
    feat[idx] = f2bf(v);
}

// ---------------------------------------------------------------- WMMA GEMM with async-LDS-staged B tiles
// Block = 256 threads = 8 waves; computes a 128-row x 64-col output tile.
// Wave w handles rows [mtg*128 + w*16, +16). B tile (32K x 64N, 4KB packed)
// is shared by all waves, double-buffered in LDS via async global->LDS DMA.
__global__ __launch_bounds__(256) void gemm_wmma_kernel(const unsigned short* __restrict__ A,
                                                        const unsigned short* __restrict__ Bp,
                                                        float* __restrict__ C) {
    __shared__ __align__(16) unsigned short sB[2][4 * 32 * 16];   // 2 x 4KB

    int nt64 = blockIdx.x % NT64N;
    int mtg  = blockIdx.x / NT64N;
    int tid  = threadIdx.x;
    int wave = tid >> 5;
    int lane = tid & 31;
    int mt   = mtg * 8 + wave;

    // A fragment pointer: row = mt*16 + lane&15 ; lanes>=16 read K offset +16 (contiguous 16)
    const unsigned short* Arow =
        A + (size_t)(mt * 16 + (lane & 15)) * KPAD + ((lane >> 4) << 4);

    const unsigned short* Btile0 = Bp + (size_t)(nt64 * KTILES) * 2048;  // 2048 bf16 = 4KB per kt

    // prologue: stage kt=0
    async_copy_b128(Btile0 + (size_t)tid * 8, &sB[0][tid * 8]);

    v8f acc[4] = {};
    for (int kt = 0; kt < KTILES; ++kt) {
        wait_async_all();
        __syncthreads();
        if (kt + 1 < KTILES)
            async_copy_b128(Btile0 + (size_t)(kt + 1) * 2048 + tid * 8,
                            &sB[(kt + 1) & 1][tid * 8]);

        ABFrag a;
        a.u[0] = *(const u32x4*)(Arow + kt * 32);
        a.u[1] = *(const u32x4*)(Arow + kt * 32 + 8);
        if (kt + 1 < KTILES) __builtin_prefetch(Arow + (kt + 1) * 32, 0, 3);

        const unsigned short* lb = &sB[kt & 1][0];
        #pragma unroll
        for (int j = 0; j < 4; ++j) {
            ABFrag b;
            const unsigned short* bp = lb + (j * 32 + lane) * 16;
            b.u[0] = *(const u32x4*)(bp);
            b.u[1] = *(const u32x4*)(bp + 8);
            acc[j] = __builtin_amdgcn_wmma_f32_16x16x32_bf16(
                false, a.v, false, b.v, (short)0, acc[j], false, false);
        }
    }

    // C layout: VGPR r -> row mt*16 + (lane<16 ? r : r+8), col = strip + lane&15
    int colBase = nt64 * 64 + (lane & 15);
    int rowBase = mt * 16 + ((lane >> 4) << 3);
    #pragma unroll
    for (int j = 0; j < 4; ++j) {
        float* cp = C + (size_t)rowBase * CN + colBase + j * 16;
        #pragma unroll
        for (int r = 0; r < 8; ++r) cp[(size_t)r * CN] = acc[j][r];
    }
}

// ---------------------------------------------------------------- per-(edge,head) logits + segment max
__global__ __launch_bounds__(256) void edge_logits_kernel(const long long* __restrict__ ei,
                                                          const float* __restrict__ xl,
                                                          const float* __restrict__ xr,
                                                          const float* __restrict__ att,
                                                          float* __restrict__ logits,
                                                          float* __restrict__ segmax) {
    int wid  = blockIdx.x * (blockDim.x >> 5) + (threadIdx.x >> 5);
    int lane = threadIdx.x & 31;
    int e = wid / H_HEADS;
    int h = wid - e * H_HEADS;
    if (e >= ETOT) return;
    int src, dst;
    if (e < E_EDGES) { src = (int)ei[e]; dst = (int)ei[E_EDGES + e]; }
    else             { src = e - E_EDGES; dst = src; }
    const float* pl = xl + (size_t)src * CN + h * C_CH;
    const float* pr = xr + (size_t)dst * CN + h * C_CH;
    const float* pa = att + h * C_CH;
    float s = 0.f;
    for (int c = lane; c < C_CH; c += 32) {
        float v = pl[c] + pr[c];
        v = (v > 0.f) ? v : LRELU * v;
        s += v * pa[c];
    }
    #pragma unroll
    for (int off = 16; off > 0; off >>= 1) s += __shfl_xor(s, off, 32);
    if (lane == 0) {
        logits[(size_t)e * H_HEADS + h] = s;
        atomicMaxFloat(&segmax[dst * H_HEADS + h], s);
    }
}

// ---------------------------------------------------------------- exp(logit - max) + segment sum (a stored in place)
__global__ void edge_alpha_kernel(const long long* __restrict__ ei,
                                  float* __restrict__ logits,
                                  const float* __restrict__ segmax,
                                  float* __restrict__ segsum) {
    int idx = blockIdx.x * blockDim.x + threadIdx.x;
    if (idx >= ETOT * H_HEADS) return;
    int e = idx / H_HEADS;
    int h = idx - e * H_HEADS;
    int dst = (e < E_EDGES) ? (int)ei[E_EDGES + e] : (e - E_EDGES);
    float m = segmax[dst * H_HEADS + h];
    if (!(m > -3.0e38f && m < 3.0e38f)) m = 0.f;   // isfinite guard
    float a = __expf(logits[idx] - m);
    logits[idx] = a;
    atomicAdd(&segsum[dst * H_HEADS + h], a);
}

// ---------------------------------------------------------------- msg aggregation, heads folded, mean over heads
__global__ __launch_bounds__(512) void aggregate_kernel(const long long* __restrict__ ei,
                                                        const float* __restrict__ xl,
                                                        const float* __restrict__ abuf,
                                                        const float* __restrict__ segsum,
                                                        float* __restrict__ agg) {
    __shared__ float s_alpha[H_HEADS];
    int e = blockIdx.x;
    int src, dst;
    if (e < E_EDGES) { src = (int)ei[e]; dst = (int)ei[E_EDGES + e]; }
    else             { src = e - E_EDGES; dst = src; }
    if (threadIdx.x < H_HEADS) {
        float a = abuf[(size_t)e * H_HEADS + threadIdx.x];
        float d = segsum[dst * H_HEADS + threadIdx.x] + 1e-16f;
        s_alpha[threadIdx.x] = a / d;
    }
    __syncthreads();
    int c = threadIdx.x;
    const float* p = xl + (size_t)src * CN + c;
    float v = 0.f;
    #pragma unroll
    for (int h = 0; h < H_HEADS; ++h) v += p[h * C_CH] * s_alpha[h];
    atomicAdd(&agg[(size_t)dst * C_CH + c], v * (1.0f / H_HEADS));
}

// ---------------------------------------------------------------- hidden = selu(agg + bias)
__global__ void update_hidden_kernel(const float* __restrict__ agg,
                                     const float* __restrict__ gat_bias,
                                     float* __restrict__ hidden) {
    int idx = blockIdx.x * blockDim.x + threadIdx.x;
    if (idx >= N_NODES * C_CH) return;
    int c = idx & (C_CH - 1);
    hidden[idx] = selu_f(agg[idx] + gat_bias[c]);
}

// ---------------------------------------------------------------- coord = clamp(hidden @ W_coord + b)
__global__ __launch_bounds__(256) void coord_update_kernel(const float* __restrict__ hidden,
                                                           const float* __restrict__ W_coord,
                                                           const float* __restrict__ b_coord,
                                                           float* __restrict__ coord) {
    int wid  = blockIdx.x * (blockDim.x >> 5) + (threadIdx.x >> 5);
    int lane = threadIdx.x & 31;
    if (wid >= N_NODES) return;
    const float* hp = hidden + (size_t)wid * C_CH;
    float sx = 0.f, sy = 0.f;
    for (int c = lane; c < C_CH; c += 32) {
        float h = hp[c];
        sx += h * W_coord[c * 2 + 0];
        sy += h * W_coord[c * 2 + 1];
    }
    #pragma unroll
    for (int off = 16; off > 0; off >>= 1) {
        sx += __shfl_xor(sx, off, 32);
        sy += __shfl_xor(sy, off, 32);
    }
    if (lane == 0) {
        float ox = coord[wid * 2 + 0], oy = coord[wid * 2 + 1];
        bool up = (ox == 1.f), down = (ox == 0.f);
        bool left = (oy == 0.f), right = (oy == 1.f);
        float nx = sx + b_coord[0];
        float ny = sy + b_coord[1];
        nx = down  ? 0.f : (up   ? 1.f : nx);
        ny = right ? 1.f : (left ? 0.f : ny);
        coord[wid * 2 + 0] = nx;
        coord[wid * 2 + 1] = ny;
    }
}

// ================================================================ launch
extern "C" void kernel_launch(void* const* d_in, const int* in_sizes, int n_in,
                              void* d_out, int out_size, void* d_ws, size_t ws_size,
                              hipStream_t stream) {
    const float*     x          = (const float*)d_in[0];
    const float*     local_feat = (const float*)d_in[1];
    const float*     conv_feat  = (const float*)d_in[2];
    const float*     W_lin      = (const float*)d_in[3];
    const float*     b_lin      = (const float*)d_in[4];
    const float*     Wl         = (const float*)d_in[5];
    const float*     Wr         = (const float*)d_in[6];
    const float*     att        = (const float*)d_in[7];
    const float*     gat_bias   = (const float*)d_in[8];
    const float*     W_coord    = (const float*)d_in[9];
    const float*     b_coord    = (const float*)d_in[10];
    const long long* edge_index = (const long long*)d_in[11];

    float* coord = (float*)d_out;   // N x 2, live coordinate buffer

    // workspace partition (256B aligned)
    char* base = (char*)d_ws;
    size_t off = 0;
    auto take = [&](size_t bytes) {
        void* p = base + off;
        off += (bytes + 255) & ~(size_t)255;
        return p;
    };
    unsigned short* feat   = (unsigned short*)take((size_t)N_NODES * KPAD * 2);
    unsigned short* Wlp    = (unsigned short*)take((size_t)KPAD * CN * 2);
    unsigned short* Wrp    = (unsigned short*)take((size_t)KPAD * CN * 2);
    float*          xl     = (float*)take((size_t)N_NODES * CN * 4);
    float*          xr     = (float*)take((size_t)N_NODES * CN * 4);
    float*          hidden = (float*)take((size_t)N_NODES * C_CH * 4);
    float*          agg    = (float*)take((size_t)N_NODES * C_CH * 4);
    float*          logits = (float*)take((size_t)ETOT * H_HEADS * 4);
    float*          segmax = (float*)take((size_t)N_NODES * H_HEADS * 4);
    float*          segsum = (float*)take((size_t)N_NODES * H_HEADS * 4);

    const int T = 256;
    coord_init_kernel<<<(N_NODES + T - 1) / T, T, 0, stream>>>(x, coord);
    init_hidden_kernel<<<(N_NODES * C_CH + T - 1) / T, T, 0, stream>>>(
        x, local_feat, conv_feat, W_lin, b_lin, hidden);

    int packN = NT64N * KTILES * 4 * 32;
    pack_w_kernel<<<(packN + T - 1) / T, T, 0, stream>>>(Wl, Wlp);
    pack_w_kernel<<<(packN + T - 1) / T, T, 0, stream>>>(Wr, Wrp);

    int gemmBlocks = (N_NODES / 128) * NT64N;            // 64 * 48 = 3072 blocks, 8 waves each
    int logitWaves = ETOT * H_HEADS;

    for (int it = 0; it < 3; ++it) {
        build_feat_kernel<<<(N_NODES * KPAD + T - 1) / T, T, 0, stream>>>(coord, hidden, feat);

        gemm_wmma_kernel<<<gemmBlocks, 256, 0, stream>>>(feat, Wlp, xl);
        gemm_wmma_kernel<<<gemmBlocks, 256, 0, stream>>>(feat, Wrp, xr);

        fill_f32_kernel<<<(N_NODES * H_HEADS + T - 1) / T, T, 0, stream>>>(
            segmax, -INFINITY, N_NODES * H_HEADS);
        fill_f32_kernel<<<(N_NODES * H_HEADS + T - 1) / T, T, 0, stream>>>(
            segsum, 0.f, N_NODES * H_HEADS);
        fill_f32_kernel<<<(N_NODES * C_CH + T - 1) / T, T, 0, stream>>>(
            agg, 0.f, N_NODES * C_CH);

        edge_logits_kernel<<<(logitWaves + 7) / 8, 256, 0, stream>>>(
            edge_index, xl, xr, att, logits, segmax);
        edge_alpha_kernel<<<(logitWaves + T - 1) / T, T, 0, stream>>>(
            edge_index, logits, segmax, segsum);
        aggregate_kernel<<<ETOT, 512, 0, stream>>>(edge_index, xl, logits, segsum, agg);

        update_hidden_kernel<<<(N_NODES * C_CH + T - 1) / T, T, 0, stream>>>(
            agg, gat_bias, hidden);
        coord_update_kernel<<<(N_NODES + 7) / 8, 256, 0, stream>>>(
            hidden, W_coord, b_coord, coord);
    }
}
